// VAMetric_65764539236772
// MI455X (gfx1250) — compile-verified, matrix operand order checked by reading, not run on hardware
//
#include <hip/hip_runtime.h>
#include <math.h>

#define DI __device__ __forceinline__

typedef __attribute__((ext_vector_type(16))) __bf16    v16bf;
typedef __attribute__((ext_vector_type(8)))  float     v8f;
typedef __attribute__((ext_vector_type(8)))  unsigned  v8u;

// ---------- bf16 helpers (round-to-nearest-even) ----------
DI unsigned short bfbits(float x) {
  unsigned u = __builtin_bit_cast(unsigned, x);
  unsigned r = u + 0x7FFFu + ((u >> 16) & 1u);
  return (unsigned short)(r >> 16);
}
DI __bf16 f2bf(float x) { return __builtin_bit_cast(__bf16, bfbits(x)); }
DI unsigned packbf2(float x, float y) {
  return (unsigned)bfbits(x) | ((unsigned)bfbits(y) << 16);
}

// K offset inside a 16x32 bf16 A/B tile for VGPR v, lane-half h.
// Doc layout: lanes 0-15 hold K {0..7, 16..23}; lanes 16-31 hold K {8..15, 24..31}.
DI int kq_of(int v, int half) { return (v < 4 ? 2 * v : 8 + 2 * v) + 8 * half; }

DI float sigmoidf_(float x) { return 1.f / (1.f + __expf(-x)); }

// ---------- weight f32 -> bf16 with zero padding ----------
__global__ void cvt_pad_kernel(const float* __restrict__ src, int srows, int scols,
                               __bf16* __restrict__ dst, int drows, int dcols) {
  int i = blockIdx.x * blockDim.x + threadIdx.x;
  int total = drows * dcols;
  if (i >= total) return;
  int r = i / dcols, ccol = i - r * dcols;
  float v = (r < srows && ccol < scols) ? src[(long)r * scols + ccol] : 0.f;
  dst[i] = f2bf(v);
}

// ---------- Phase 1: conv (1152 -> 300 per timestep) as bf16 WMMA GEMM ----------
// Each wave: one 16-row M tile x 10 N tiles (half of padded N=320). K = 36 steps of 32.
__global__ void __launch_bounds__(256)
conv_gemm_kernel(const float* __restrict__ vfeat, const float* __restrict__ afeat,
                 const __bf16* __restrict__ convw,   // [320, 1152] bf16, rows>=300 zero
                 const float* __restrict__ convb,    // [300]
                 __bf16* __restrict__ feat1) {       // [131072, 320] bf16, cols>=300 zero
  const int lane = threadIdx.x & 31;
  const int wv   = threadIdx.x >> 5;
  const int half = lane >> 4;
  const int idx  = lane & 15;
  const int w    = blockIdx.x * 8 + wv;       // 0..16383
  const int mt    = w >> 1;                   // 0..8191
  const int nbase = (w & 1) * 160;            // 0 or 160
  const long arow = (long)mt * 16 + idx;

  v8f acc[10] = {};
  for (int ks = 0; ks < 36; ++ks) {
    const int kbase = ks * 32;  // never straddles the 1024 boundary
    const float* asrc = (kbase < 1024) ? (vfeat + arow * 1024 + kbase)
                                       : (afeat + arow * 128 + (kbase - 1024));
    v8u au;
#pragma unroll
    for (int v = 0; v < 8; ++v) {
      float2 t = *(const float2*)(asrc + kq_of(v, half));
      au[v] = packbf2(t.x, t.y);
    }
    v16bf a = __builtin_bit_cast(v16bf, au);
#pragma unroll
    for (int nt = 0; nt < 10; ++nt) {
      const __bf16* wr = convw + (long)(nbase + nt * 16 + idx) * 1152 + kbase;
      v8u bu;
#pragma unroll
      for (int v = 0; v < 8; ++v) bu[v] = *(const unsigned*)(wr + kq_of(v, half));
      acc[nt] = __builtin_amdgcn_wmma_f32_16x16x32_bf16(
          false, a, false, __builtin_bit_cast(v16bf, bu), (short)0, acc[nt], false, false);
    }
  }
  // Epilogue: bias + ReLU, store bf16; pad columns (>=300) get exact zero.
  const long orow = (long)mt * 16 + 8 * half;
#pragma unroll
  for (int nt = 0; nt < 10; ++nt) {
    const int n = nbase + nt * 16 + idx;
    const float bias = (n < 300) ? convb[n] : 0.f;
#pragma unroll
    for (int r = 0; r < 8; ++r) {
      float val = 0.f;
      if (n < 300) {
        val = acc[nt][r] + bias;
        val = val > 0.f ? val : 0.f;
      }
      feat1[(orow + r) * 320 + n] = f2bf(val);
    }
  }
}

// ---------- fused LSTM cell: 4-gate WMMA + in-register nonlinearity ----------
DI void quad_step(v8f (&acc)[4], const __bf16* __restrict__ xrow,
                  const __bf16* __restrict__ w0, const __bf16* __restrict__ w1,
                  const __bf16* __restrict__ w2, const __bf16* __restrict__ w3,
                  int kbase, int half) {
  v8u au, b0u, b1u, b2u, b3u;
#pragma unroll
  for (int v = 0; v < 8; ++v) {
    const int k = kbase + kq_of(v, half);
    au[v]  = *(const unsigned*)(xrow + k);
    b0u[v] = *(const unsigned*)(w0 + k);
    b1u[v] = *(const unsigned*)(w1 + k);
    b2u[v] = *(const unsigned*)(w2 + k);
    b3u[v] = *(const unsigned*)(w3 + k);
  }
  v16bf a = __builtin_bit_cast(v16bf, au);
  acc[0] = __builtin_amdgcn_wmma_f32_16x16x32_bf16(false, a, false, __builtin_bit_cast(v16bf, b0u), (short)0, acc[0], false, false);
  acc[1] = __builtin_amdgcn_wmma_f32_16x16x32_bf16(false, a, false, __builtin_bit_cast(v16bf, b1u), (short)0, acc[1], false, false);
  acc[2] = __builtin_amdgcn_wmma_f32_16x16x32_bf16(false, a, false, __builtin_bit_cast(v16bf, b2u), (short)0, acc[2], false, false);
  acc[3] = __builtin_amdgcn_wmma_f32_16x16x32_bf16(false, a, false, __builtin_bit_cast(v16bf, b3u), (short)0, acc[3], false, false);
}

// One wave = 16 batch rows x 16 hidden cols; accumulates gates i,f,g,o at
// columns n0, n0+128, n0+256, n0+384 of the [1024,512] gate matrix.
__global__ void __launch_bounds__(256)
lstm_cell_kernel(const __bf16* __restrict__ x, long xstride, int kpad,   // x: [1024, kpad]-ish rows @ stride
                 const __bf16* __restrict__ hrec,                         // [1024,128] bf16 (ping)
                 const __bf16* __restrict__ w_ih,                         // [512, kpad] bf16 (padded)
                 const __bf16* __restrict__ w_hh,                         // [512, 128] bf16
                 const float* __restrict__ b_ih, const float* __restrict__ b_hh,
                 float* __restrict__ c,                                   // [1024,128] f32 state
                 float* __restrict__ hf,                                  // [1024,128] f32 h out
                 __bf16* __restrict__ hb) {                               // [1024,128] bf16 h out (pong)
  const int lane = threadIdx.x & 31;
  const int wv   = threadIdx.x >> 5;
  const int half = lane >> 4;
  const int idx  = lane & 15;
  const int w  = blockIdx.x * 8 + wv;    // 0..511
  const int m0 = (w >> 3) * 16;          // batch tile
  const int n0 = (w & 7) * 16;           // hidden-col tile

  v8f acc[4] = {};
  {  // x @ w_ih^T
    const __bf16* xrow = x + (long)(m0 + idx) * xstride;
    const __bf16* w0 = w_ih + (long)(n0 + idx) * kpad;
    const __bf16* w1 = w0 + (long)128 * kpad;
    const __bf16* w2 = w1 + (long)128 * kpad;
    const __bf16* w3 = w2 + (long)128 * kpad;
    const int nks = kpad >> 5;
    for (int ks = 0; ks < nks; ++ks) quad_step(acc, xrow, w0, w1, w2, w3, ks * 32, half);
  }
  {  // h @ w_hh^T
    const __bf16* xrow = hrec + (long)(m0 + idx) * 128;
    const __bf16* w0 = w_hh + (long)(n0 + idx) * 128;
    const __bf16* w1 = w0 + 128 * 128;
    const __bf16* w2 = w1 + 128 * 128;
    const __bf16* w3 = w2 + 128 * 128;
    for (int ks = 0; ks < 4; ++ks) quad_step(acc, xrow, w0, w1, w2, w3, ks * 32, half);
  }
  // Elementwise: gates live per-lane-aligned across the 4 accumulators.
  const int j = n0 + idx;
  const float bi = b_ih[j]       + b_hh[j];
  const float bF = b_ih[j + 128] + b_hh[j + 128];
  const float bG = b_ih[j + 256] + b_hh[j + 256];
  const float bO = b_ih[j + 384] + b_hh[j + 384];
#pragma unroll
  for (int r = 0; r < 8; ++r) {
    const int off = (m0 + r + 8 * half) * 128 + j;
    const float ii = sigmoidf_(acc[0][r] + bi);
    const float ff = sigmoidf_(acc[1][r] + bF);
    const float gg = tanhf(acc[2][r] + bG);
    const float oo = sigmoidf_(acc[3][r] + bO);
    const float cn = ff * c[off] + ii * gg;
    c[off] = cn;
    const float hn = oo * tanhf(cn);
    hf[off] = hn;
    hb[off] = f2bf(hn);
  }
}

// ---------- final FC [1024,128] -> [1024,1] + ReLU ----------
__global__ void fc_kernel(const float* __restrict__ h2, const float* __restrict__ fc_w,
                          const float* __restrict__ fc_b, float* __restrict__ out) {
  int b = blockIdx.x * blockDim.x + threadIdx.x;
  if (b >= 1024) return;
  float s = fc_b[0];
#pragma unroll 4
  for (int j = 0; j < 128; ++j) s += h2[b * 128 + j] * fc_w[j];
  out[b] = s > 0.f ? s : 0.f;
}

extern "C" void kernel_launch(void* const* d_in, const int* in_sizes, int n_in,
                              void* d_out, int out_size, void* d_ws, size_t ws_size,
                              hipStream_t stream) {
  (void)in_sizes; (void)n_in; (void)out_size; (void)ws_size;
  const float* vfeat  = (const float*)d_in[0];
  const float* afeat  = (const float*)d_in[1];
  const float* conv_w = (const float*)d_in[2];
  const float* conv_b = (const float*)d_in[3];
  const float* w_ih1  = (const float*)d_in[4];
  const float* w_hh1  = (const float*)d_in[5];
  const float* b_ih1  = (const float*)d_in[6];
  const float* b_hh1  = (const float*)d_in[7];
  const float* w_ih2  = (const float*)d_in[8];
  const float* w_hh2  = (const float*)d_in[9];
  const float* b_ih2  = (const float*)d_in[10];
  const float* b_hh2  = (const float*)d_in[11];
  const float* fc_w   = (const float*)d_in[12];
  const float* fc_b   = (const float*)d_in[13];
  float* out = (float*)d_out;

  // ---- workspace carve-out (~88.5 MB) ----
  char* ws = (char*)d_ws;
  size_t off = 0;
  auto carve = [&](size_t bytes) -> char* {
    char* p = ws + off;
    off += (bytes + 255) & ~(size_t)255;
    return p;
  };
  __bf16* feat1    = (__bf16*)carve(131072ull * 320 * 2);  // [B*T, 320]
  __bf16* convw_bf = (__bf16*)carve(320ull * 1152 * 2);    // [320, 1152]
  __bf16* wih1_bf  = (__bf16*)carve(512ull * 320 * 2);     // [512, 320]
  __bf16* whh1_bf  = (__bf16*)carve(512ull * 128 * 2);
  __bf16* wih2_bf  = (__bf16*)carve(512ull * 128 * 2);
  __bf16* whh2_bf  = (__bf16*)carve(512ull * 128 * 2);
  float*  c1  = (float*)carve(1024ull * 128 * 4);
  float*  c2  = (float*)carve(1024ull * 128 * 4);
  float*  h1f = (float*)carve(1024ull * 128 * 4);
  float*  h2f = (float*)carve(1024ull * 128 * 4);
  __bf16* h1b = (__bf16*)carve(2ull * 1024 * 128 * 2);     // ping-pong
  __bf16* h2b = (__bf16*)carve(2ull * 1024 * 128 * 2);     // ping-pong

  // ---- weight conversion (padded bf16) ----
  cvt_pad_kernel<<<(320 * 1152 + 255) / 256, 256, 0, stream>>>(conv_w, 300, 1152, convw_bf, 320, 1152);
  cvt_pad_kernel<<<(512 * 320 + 255) / 256, 256, 0, stream>>>(w_ih1, 512, 300, wih1_bf, 512, 320);
  cvt_pad_kernel<<<(512 * 128 + 255) / 256, 256, 0, stream>>>(w_hh1, 512, 128, whh1_bf, 512, 128);
  cvt_pad_kernel<<<(512 * 128 + 255) / 256, 256, 0, stream>>>(w_ih2, 512, 128, wih2_bf, 512, 128);
  cvt_pad_kernel<<<(512 * 128 + 255) / 256, 256, 0, stream>>>(w_hh2, 512, 128, whh2_bf, 512, 128);

  // ---- phase 1: big GEMM (memory-bound: 576 MB activations @ 23.3 TB/s) ----
  conv_gemm_kernel<<<2048, 256, 0, stream>>>(vfeat, afeat, convw_bf, conv_b, feat1);

  // ---- state init (ws is poisoned; must zero every call) ----
  hipMemsetAsync(c1, 0, 1024ull * 128 * 4, stream);
  hipMemsetAsync(c2, 0, 1024ull * 128 * 4, stream);
  hipMemsetAsync(h1b, 0, 2ull * 1024 * 128 * 2, stream);
  hipMemsetAsync(h2b, 0, 2ull * 1024 * 128 * 2, stream);

  // ---- phase 2: sequential 2-layer LSTM, one fused kernel per cell per step ----
  for (int t = 0; t < 128; ++t) {
    const int p = t & 1;
    // cell 1: x_t = feat1[:, t, :]  (row b at offset (b*128+t)*320)
    lstm_cell_kernel<<<64, 256, 0, stream>>>(
        feat1 + (long)t * 320, (long)128 * 320, 320,
        h1b + (size_t)p * 131072, wih1_bf, whh1_bf, b_ih1, b_hh1,
        c1, h1f, h1b + (size_t)(1 - p) * 131072);
    // cell 2: x = h1 (just produced)
    lstm_cell_kernel<<<64, 256, 0, stream>>>(
        h1b + (size_t)(1 - p) * 131072, 128, 128,
        h2b + (size_t)p * 131072, wih2_bf, whh2_bf, b_ih2, b_hh2,
        c2, h2f, h2b + (size_t)(1 - p) * 131072);
  }

  // ---- phase 3: FC + ReLU ----
  fc_kernel<<<4, 256, 0, stream>>>(h2f, fc_w, fc_b, out);
}
